// Network_47605417508892
// MI455X (gfx1250) — matrix operand order repaired and checked by reading, not compiled, not run on hardware
//
#include <hip/hip_runtime.h>
#include <math.h>

// ---------------------------------------------------------------------------
// MI455X (gfx1250, wave32) implementation.
// Dense GEMMs -> v_wmma_f32_16x16x32_f16, 16x64 per wave (A-frag reuse, 4 wmma
// per K-step), W slab staged block-wide through LDS with
// global_load_async_to_lds_b128 + s_wait_asynccnt double buffering.
// Attention msg_in broadcast factored into hidden-GEMM + 49-entry PE tables.
// ---------------------------------------------------------------------------

typedef _Float16 f16;
typedef __attribute__((ext_vector_type(16))) _Float16 v16h;
typedef __attribute__((ext_vector_type(8)))  _Float16 v8h;
typedef __attribute__((ext_vector_type(8)))  float    v8f;

#define BB   8
#define NN   128
#define BN   1024
#define HD   256
#define MSG  128
#define NH   4
#define MD   32
#define PE   16
#define FOV  7
#define RAD  3
#define CIN  8
#define SLABW 40   // padded LDS row stride (f16) for 32-K slab, spreads banks

// ---------------------------------------------------------------------------
// Generic f32 -> f16 convert with row extraction / zero K-padding.
__global__ void k_cvt(const float* __restrict__ src, f16* __restrict__ dst,
                      int rows, int srcK, int srcLd, int dstK) {
  int i = blockIdx.x * blockDim.x + threadIdx.x;
  if (i >= rows * dstK) return;
  int r = i / dstK, c = i - r * dstK;
  f16 v = (f16)0.0f;
  if (c < srcK) v = (f16)src[(size_t)r * srcLd + c];
  dst[i] = v;
}

// im2col for conv1: obs (BN,8,7,7) f32 -> col (BN*25, 96) f16 (K=72 padded)
__global__ void k_im2col1(const float* __restrict__ obs, f16* __restrict__ col) {
  int i = blockIdx.x * blockDim.x + threadIdx.x;
  if (i >= BN * 25 * 96) return;
  int row = i / 96, c = i - row * 96;
  int n = row / 25, p = row - n * 25, oy = p / 5, ox = p - oy * 5;
  f16 v = (f16)0.0f;
  if (c < 72) {
    int ch = c / 9, k = c - ch * 9, ky = k / 3, kx = k - ky * 3;
    v = (f16)obs[((size_t)(n * CIN + ch) * 7 + (oy + ky)) * 7 + (ox + kx)];
  }
  col[i] = v;
}

// im2col for conv2: act1 NHWC f16 (BN,5,5,64) -> col (BN*9, 576) f16
__global__ void k_im2col2(const f16* __restrict__ act1, f16* __restrict__ col) {
  int i = blockIdx.x * blockDim.x + threadIdx.x;
  if (i >= BN * 9 * 576) return;
  int row = i / 576, c = i - row * 576;
  int n = row / 9, p = row - n * 9, oy = p / 3, ox = p - oy * 3;
  int ch = c / 9, k = c - ch * 9, ky = k / 3, kx = k - ky * 3;
  col[i] = act1[((size_t)(n * 5 + (oy + ky)) * 5 + (ox + kx)) * 64 + ch];
}

// "im2col" for conv3 (1x1 output): act2 NHWC f16 (BN,3,3,128) -> col (BN,1152)
__global__ void k_im2col3(const f16* __restrict__ act2, f16* __restrict__ col) {
  int i = blockIdx.x * blockDim.x + threadIdx.x;
  if (i >= BN * 1152) return;
  int n = i / 1152, c = i - n * 1152;
  int ch = c / 9, k = c - ch * 9, y = k / 3, x = k - y * 3;
  col[i] = act2[((size_t)(n * 3 + y) * 3 + x) * 128 + ch];
}

// ---------------------------------------------------------------------------
// gfx1250 async global->LDS copy (16B per lane) + ASYNCcnt waits.
__device__ __forceinline__ void async_b128(unsigned ldsoff, const void* g) {
  asm volatile("global_load_async_to_lds_b128 %0, %1, off"
               :: "v"(ldsoff), "v"(g) : "memory");
}
__device__ __forceinline__ void wait_async1() {
  asm volatile("s_wait_asynccnt 0x1" ::: "memory");
}
__device__ __forceinline__ void wait_async0() {
  asm volatile("s_wait_asynccnt 0x0" ::: "memory");
}

// ---------------------------------------------------------------------------
// WMMA GEMM: C[m,n] = act( alpha * sum_k A[m,k]*W[n,k] + bias[n] )
// A (M,K) f16 row-major (lda), W (N,K) f16 row-major (ldw). K mult of 32,
// M mult of 128, N mult of 64. Block: 256 thr = 8 waves, computes 128x64 of C;
// wave w -> rows m0=blk*128+16w, all 4 n-subtiles (A frag reused 4x).
// W slab (64 rows x 32 K) double-buffered in LDS via async loads: each wave
// issues one global_load_async_to_lds_b128 covering 8 rows per K-step.
// Optional batching over blockIdx.z: off = (z/bMod)*s0 + (z%bMod)*s1.
// Fragment layout (CDNA5 ISA):
//   A: lane l -> row (l&15), K halves at {ka, ka+16}, ka=(l>>4)*8
//   B: lane l -> col (l&15), K halves at {kb, kb+16}, kb=(l>>4)*8
//   C: vgpr q -> row m0+q+8*(l>>4), col n0+(l&15)
__global__ __launch_bounds__(256) void k_gemm(
    const f16* __restrict__ A, const f16* __restrict__ W,
    const float* __restrict__ bias,
    float* __restrict__ C32, f16* __restrict__ C16,
    int M, int N, int Kp, int lda, int ldw, int ldc32, int ldc16,
    float alpha, int leaky,
    long long bA0, long long bA1, long long bB0, long long bB1,
    long long bC0, long long bC1, int bMod) {
  __shared__ f16 slab[2][64 * SLABW];
  const int wave = threadIdx.x >> 5;
  const int lane = threadIdx.x & 31;
  const int tilesM = M >> 7;
  const int m0 = (blockIdx.x % tilesM) * 128 + wave * 16;
  const int n0 = (blockIdx.x / tilesM) * 64;
  const long long zo = (long long)(blockIdx.z / bMod);
  const long long zi = (long long)(blockIdx.z % bMod);
  A += zo * bA0 + zi * bA1;
  W += zo * bB0 + zi * bB1;
  const long long coff = zo * bC0 + zi * bC1;

  // async prefetch mapping: this wave stages rows 8w..8w+7 of the W slab
  const int prow = (wave << 3) + (lane >> 2);   // 0..63
  const int pcol = (lane & 3) << 3;             // f16 elems {0,8,16,24}
  const f16* wsrc = W + (size_t)(n0 + prow) * ldw + pcol;
  const unsigned lds0 = (unsigned)(size_t)&slab[0][prow * SLABW + pcol];
  const unsigned lds1 = (unsigned)(size_t)&slab[1][prow * SLABW + pcol];

  const int r  = lane & 15;
  const int ka = (lane >> 4) << 3;   // A K-offset {0,8}
  const int kb = (lane >> 4) << 3;   // B K-offset {0,8}
  const f16* ap = A + (size_t)(m0 + r) * lda + ka;

  v8f acc[4] = {{}, {}, {}, {}};
  const int nk = Kp >> 5;
  async_b128(lds0, wsrc);                       // prefetch slab 0
  for (int s = 0; s < nk; ++s) {
    __syncthreads();                            // readers of buf[(s+1)&1] done
    if (s + 1 < nk) {
      async_b128(((s + 1) & 1) ? lds1 : lds0, wsrc + (size_t)(s + 1) * 32);
      wait_async1();                            // slab s landed (in-order)
    } else {
      wait_async0();
    }
    __syncthreads();                            // all waves' slab-s writes done
    union { v16h v; v8h h[2]; } af;
    af.h[0] = *(const v8h*)(ap + s * 32);
    af.h[1] = *(const v8h*)(ap + s * 32 + 16);
    const f16* sb = &slab[s & 1][0];
#pragma unroll
    for (int v = 0; v < 4; ++v) {
      union { v16h v; v8h h[2]; } bf;
      const f16* bp = sb + ((v << 4) + r) * SLABW + kb;
      bf.h[0] = *(const v8h*)bp;
      bf.h[1] = *(const v8h*)(bp + 16);
      acc[v] = __builtin_amdgcn_wmma_f32_16x16x32_f16(false, af.v, false, bf.v,
                                                      (short)0, acc[v], false,
                                                      false);
    }
  }
  const int mbase = m0 + ((lane >> 4) << 3);
#pragma unroll
  for (int v = 0; v < 4; ++v) {
    const int n = n0 + (v << 4) + r;
    const float bv = bias ? bias[n] : 0.0f;
#pragma unroll
    for (int q = 0; q < 8; q++) {
      float val = alpha * acc[v][q] + bv;
      if (leaky) val = (val >= 0.0f) ? val : 0.2f * val;
      if (C32) C32[coff + (size_t)(mbase + q) * ldc32 + n] = val;
      if (C16) C16[coff + (size_t)(mbase + q) * ldc16 + n] = (f16)val;
    }
  }
}

// ---------------------------------------------------------------------------
// latent tail: cols [256,288) of latent f16 (BN,288): last_act then zeros.
__global__ void k_latent_tail(const float* __restrict__ last_act,
                              f16* __restrict__ lat) {
  int i = blockIdx.x * blockDim.x + threadIdx.x;
  if (i >= BN * 32) return;
  int row = i / 32, c = i - row * 32;
  f16 v = (f16)0.0f;
  if (c < 5) v = (f16)last_act[row * 5 + c];
  lat[(size_t)row * 288 + 256 + c] = v;
}

// Fused GRU elementwise. gi,gh (M,3*D); out = mask?gru:hprev.
__global__ void k_gru(const float* __restrict__ gi, const float* __restrict__ gh,
                      const float* __restrict__ hprev,
                      const unsigned char* __restrict__ hasmsg,
                      float* __restrict__ h32, f16* __restrict__ h16, int M) {
  int i = blockIdx.x * blockDim.x + threadIdx.x;
  if (i >= M * HD) return;
  int row = i / HD, j = i - row * HD;
  size_t o = (size_t)row * (3 * HD);
  float ir = gi[o + j], iz = gi[o + HD + j], in_ = gi[o + 2 * HD + j];
  float hr = gh[o + j], hz = gh[o + HD + j], hn_ = gh[o + 2 * HD + j];
  float rr = 1.0f / (1.0f + __expf(-(ir + hr)));
  float zz = 1.0f / (1.0f + __expf(-(iz + hz)));
  float nn = tanhf(in_ + rr * hn_);
  float hp = hprev[i];
  float out = (1.0f - zz) * nn + zz * hp;
  if (hasmsg && !hasmsg[row]) out = hp;
  h32[i] = out;
  h16[i] = (f16)out;
}

// has_msg per (b,j): OR over senders i of mask (transp: use cm[b][j][i]).
__global__ void k_hasmsg(const unsigned char* __restrict__ cm,
                         unsigned char* __restrict__ hm, int transp) {
  int t = blockIdx.x * blockDim.x + threadIdx.x;
  if (t >= BN) return;
  int b = t >> 7, j = t & 127;
  const unsigned char* base = cm + (size_t)b * NN * NN;
  int any = 0;
  for (int i = 0; i < NN; i++)
    any |= transp ? base[j * NN + i] : base[i * NN + j];
  hm[t] = (unsigned char)(any ? 1 : 0);
}

// LayerNorm: wave per row, writes f16.
__global__ void k_layernorm(const float* __restrict__ x, const float* __restrict__ s,
                            const float* __restrict__ bp, f16* __restrict__ y16,
                            int M) {
  int row = blockIdx.x * (blockDim.x >> 5) + (threadIdx.x >> 5);
  if (row >= M) return;
  int lane = threadIdx.x & 31;
  const float* xr = x + (size_t)row * HD;
  float sum = 0.0f, sq = 0.0f;
  for (int c = lane; c < HD; c += 32) { float v = xr[c]; sum += v; sq += v * v; }
  for (int o = 16; o; o >>= 1) { sum += __shfl_xor(sum, o, 32); sq += __shfl_xor(sq, o, 32); }
  float mean = sum * (1.0f / HD);
  float inv = rsqrtf(sq * (1.0f / HD) - mean * mean + 1e-5f);
  for (int c = lane; c < HD; c += 32)
    y16[(size_t)row * HD + c] = (f16)((xr[c] - mean) * inv * s[c] + bp[c]);
}

// PE table: table[e][o] = sum_p W[o][256+p] * (posW[p][e] + posb[p]),  (49,128)
__global__ void k_petable(const float* __restrict__ W, const float* __restrict__ posW,
                          const float* __restrict__ posb, float* __restrict__ table) {
  int t = blockIdx.x * blockDim.x + threadIdx.x;
  if (t >= 49 * MSG) return;
  int e = t / MSG, o = t - e * MSG;
  float acc = 0.0f;
  for (int p = 0; p < PE; p++)
    acc += W[(size_t)o * 272 + 256 + p] * (posW[p * 49 + e] + posb[p]);
  table[t] = acc;
}

// relative-pos -> PE index (same for both layers)
__global__ void k_relidx(const int* __restrict__ rp, unsigned char* __restrict__ idx) {
  int t = blockIdx.x * blockDim.x + threadIdx.x;
  if (t >= BB * NN * NN) return;
  int x = rp[2 * t], y = rp[2 * t + 1];
  int v;
  if (x < -RAD || x > RAD || y < -RAD || y > RAD) v = RAD + RAD * FOV;
  else v = (x + RAD) + (y + RAD) * FOV;
  idx[t] = (unsigned char)v;
}

// qkp[row][h][e] = scale * dot32(q[row][h*32:], kp[e][h*32:])
__global__ void k_qkp(const float* __restrict__ q, const float* __restrict__ kp,
                      float* __restrict__ qkp, float scale) {
  int t = blockIdx.x * blockDim.x + threadIdx.x;
  if (t >= BN * NH * 49) return;
  int e = t % 49, h = (t / 49) % NH, row = t / (49 * NH);
  const float* qp = q + (size_t)row * MSG + h * MD;
  const float* kpp = kp + (size_t)e * MSG + h * MD;
  float acc = 0.0f;
  for (int d = 0; d < MD; d++) acc += qp[d] * kpp[d];
  qkp[t] = acc * scale;
}

// masked softmax over senders i, in place on s (b,h,j,i). Wave per (b,h,j).
__global__ void k_softmax(float* __restrict__ s, const float* __restrict__ qkp,
                          const unsigned char* __restrict__ idx,
                          const unsigned char* __restrict__ cm, int transp) {
  int wid = blockIdx.x * (blockDim.x >> 5) + (threadIdx.x >> 5);
  if (wid >= BB * NH * NN) return;
  int lane = threadIdx.x & 31;
  int j = wid & 127, h = (wid >> 7) & 3, b = wid >> 9;
  float* srow = s + (size_t)wid * NN;
  const unsigned char* idxb = idx + (size_t)b * NN * NN;
  const unsigned char* cmb = cm + (size_t)b * NN * NN;
  const float* qk = qkp + ((size_t)(b * NN + j) * NH + h) * 49;
  float vals[4];
  float mx = -3.0e38f;
#pragma unroll
  for (int u = 0; u < 4; u++) {
    int i = lane + 32 * u;
    int m = transp ? cmb[j * NN + i] : cmb[i * NN + j];
    float v = -3.0e38f;
    if (m) v = srow[i] + qk[idxb[i * NN + j]];
    vals[u] = v;
    mx = fmaxf(mx, v);
  }
  for (int o = 16; o; o >>= 1) mx = fmaxf(mx, __shfl_xor(mx, o, 32));
  float sum = 0.0f;
#pragma unroll
  for (int u = 0; u < 4; u++) { float e = __expf(vals[u] - mx); vals[u] = e; sum += e; }
  for (int o = 16; o; o >>= 1) sum += __shfl_xor(sum, o, 32);
  float inv = 1.0f / sum;
#pragma unroll
  for (int u = 0; u < 4; u++) srow[lane + 32 * u] = vals[u] * inv;
}

// agg(b,j,h,d) = sum_i w(b,h,j,i) * (vh(b,i)[h*32+d] + vp[idx(b,i,j)][h*32+d])
__global__ void k_agg(const float* __restrict__ w, const float* __restrict__ vh,
                      const float* __restrict__ vp, const unsigned char* __restrict__ idx,
                      float* __restrict__ a32, f16* __restrict__ a16) {
  int wid = blockIdx.x * (blockDim.x >> 5) + (threadIdx.x >> 5);
  if (wid >= BB * NH * NN) return;
  int lane = threadIdx.x & 31;
  int j = wid & 127, h = (wid >> 7) & 3, b = wid >> 9;
  const float* wrow = w + (size_t)wid * NN;
  const unsigned char* idxb = idx + (size_t)b * NN * NN;
  float acc = 0.0f;
  for (int i = 0; i < NN; i++) {
    float wi = wrow[i];
    float vv = vh[(size_t)(b * NN + i) * MSG + h * MD + lane] +
               vp[(size_t)idxb[i * NN + j] * MSG + h * MD + lane];
    acc += wi * vv;
  }
  size_t o = (size_t)(b * NN + j) * MSG + h * MD + lane;
  a32[o] = acc;
  a16[o] = (f16)acc;
}

// dueling heads: wave per row. out = state + adv - mean(adv)
__global__ void k_heads(const float* __restrict__ h2, const float* __restrict__ aW,
                        const float* __restrict__ aB, const float* __restrict__ sW,
                        const float* __restrict__ sB, float* __restrict__ out) {
  int row = blockIdx.x * (blockDim.x >> 5) + (threadIdx.x >> 5);
  if (row >= BN) return;
  int lane = threadIdx.x & 31;
  const float* hr = h2 + (size_t)row * HD;
  float a[5] = {0, 0, 0, 0, 0}, st = 0.0f;
  for (int c = lane; c < HD; c += 32) {
    float v = hr[c];
#pragma unroll
    for (int o = 0; o < 5; o++) a[o] += v * aW[o * HD + c];
    st += v * sW[c];
  }
  for (int off = 16; off; off >>= 1) {
#pragma unroll
    for (int o = 0; o < 5; o++) a[o] += __shfl_xor(a[o], off, 32);
    st += __shfl_xor(st, off, 32);
  }
  if (lane == 0) {
    float mean = 0.0f;
#pragma unroll
    for (int o = 0; o < 5; o++) { a[o] += aB[o]; mean += a[o]; }
    mean *= 0.2f;
    st += sB[0];
#pragma unroll
    for (int o = 0; o < 5; o++) out[row * 5 + o] = st + a[o] - mean;
  }
}

// ---------------------------------------------------------------------------
// Host side
// ---------------------------------------------------------------------------
struct CommW {
  const float *ln_s, *ln_b, *posW, *posb, *Wk, *bk, *Wv, *bv, *Wq, *bq,
      *Wagg, *bagg, *Wih, *Whh, *bih, *bhh;
};

static inline void gemm(hipStream_t st, const f16* A, const f16* W, const float* bias,
                        float* C32, f16* C16, int M, int N, int Kp, int lda, int ldw,
                        int ldc32, int ldc16, float alpha, int leaky, int batch = 1,
                        int bMod = 1, long long bA0 = 0, long long bA1 = 0,
                        long long bB0 = 0, long long bB1 = 0, long long bC0 = 0,
                        long long bC1 = 0) {
  dim3 grid((M / 128) * (N / 64), 1, batch);
  k_gemm<<<grid, 256, 0, st>>>(A, W, bias, C32, C16, M, N, Kp, lda, ldw, ldc32,
                               ldc16, alpha, leaky, bA0, bA1, bB0, bB1, bC0, bC1,
                               bMod);
}

extern "C" void kernel_launch(void* const* d_in, const int* in_sizes, int n_in,
                              void* d_out, int out_size, void* d_ws, size_t ws_size,
                              hipStream_t stream) {
  // ---- input pytree mapping (auto-detect flatten order) -------------------
  const float *obs, *last_act, *hidden_in;
  const int* relpos;
  const unsigned char* cm;
  const float *convW[3], *convB[3];
  const float *recWih, *recWhh, *recBih, *recBhh;
  const float *advW, *advB, *stW, *stB;
  CommW L[2];  // L[0]=req, L[1]=rep
  auto F = [&](int i) { return (const float*)d_in[i]; };

  if (in_sizes[0] == BB * NN * NN) {
    // jax.tree_util sorted-key order:
    // comm_mask, hidden_in, last_act, obs, params{adv,conv,rec,rep,req,state}, relative_pos
    cm = (const unsigned char*)d_in[0];
    hidden_in = F(1); last_act = F(2); obs = F(3);
    advW = F(4); advB = F(5);
    for (int c = 0; c < 3; c++) { convW[c] = F(6 + 2 * c); convB[c] = F(7 + 2 * c); }
    recWhh = F(12); recWih = F(13); recBhh = F(14); recBih = F(15);
    for (int l = 0; l < 2; l++) {
      int o = (l == 0) ? 32 : 16;  // req at 32.., rep at 16..
      CommW& c = L[l];
      c.Wagg = F(o + 0);  c.Whh = F(o + 1);  c.Wih = F(o + 2);  c.Wk = F(o + 3);
      c.Wq = F(o + 4);    c.Wv = F(o + 5);   c.bagg = F(o + 6); c.bhh = F(o + 7);
      c.bih = F(o + 8);   c.bk = F(o + 9);   c.bq = F(o + 10);  c.bv = F(o + 11);
      c.ln_b = F(o + 12); c.ln_s = F(o + 13); c.posW = F(o + 14); c.posb = F(o + 15);
    }
    stW = F(48); stB = F(49);
    relpos = (const int*)d_in[50];
  } else {
    // insertion order: obs, last_act, hidden_in, relative_pos, comm_mask,
    // params{conv, rec{Wih,Whh,bih,bhh}, req{...}, rep{...}, adv, state}
    obs = F(0); last_act = F(1); hidden_in = F(2);
    relpos = (const int*)d_in[3];
    cm = (const unsigned char*)d_in[4];
    for (int c = 0; c < 3; c++) { convW[c] = F(5 + 2 * c); convB[c] = F(6 + 2 * c); }
    recWih = F(11); recWhh = F(12); recBih = F(13); recBhh = F(14);
    for (int l = 0; l < 2; l++) {
      int o = 15 + 16 * l;  // req then rep
      CommW& c = L[l];
      c.ln_s = F(o + 0);  c.ln_b = F(o + 1);  c.posW = F(o + 2);  c.posb = F(o + 3);
      c.Wk = F(o + 4);    c.bk = F(o + 5);    c.Wv = F(o + 6);    c.bv = F(o + 7);
      c.Wq = F(o + 8);    c.bq = F(o + 9);    c.Wagg = F(o + 10); c.bagg = F(o + 11);
      c.Wih = F(o + 12);  c.Whh = F(o + 13);  c.bih = F(o + 14);  c.bhh = F(o + 15);
    }
    advW = F(47); advB = F(48); stW = F(49); stB = F(50);
  }
  float* out = (float*)d_out;

  // ---- workspace bump allocator ------------------------------------------
  char* ws = (char*)d_ws;
  size_t off = 0;
  auto A = [&](size_t bytes) -> void* {
    void* p = ws + off;
    off = (off + bytes + 255) & ~(size_t)255;
    return p;
  };
  // persistent
  f16* wW0 = (f16*)A(64 * 96 * 2);
  f16* wW1 = (f16*)A(128 * 576 * 2);
  f16* wW2 = (f16*)A(256 * 1152 * 2);
  f16* wRih = (f16*)A(768 * 288 * 2);
  f16* wRhh = (f16*)A(768 * 256 * 2);
  f16 *wQ[2], *wKh[2], *wVh[2], *wAg[2], *wCih[2], *wChh[2];
  for (int l = 0; l < 2; l++) {
    wQ[l] = (f16*)A(128 * 256 * 2);  wKh[l] = (f16*)A(128 * 256 * 2);
    wVh[l] = (f16*)A(128 * 256 * 2); wAg[l] = (f16*)A(128 * 128 * 2);
    wCih[l] = (f16*)A(768 * 128 * 2); wChh[l] = (f16*)A(768 * 256 * 2);
  }
  f16* lat16 = (f16*)A((size_t)BN * 288 * 2);
  f16* hid16 = (f16*)A((size_t)BN * HD * 2);
  float* hA32 = (float*)A((size_t)BN * HD * 4);
  f16* hA16 = (f16*)A((size_t)BN * HD * 2);
  float* hB32 = (float*)A((size_t)BN * HD * 4);
  f16* hB16 = (f16*)A((size_t)BN * HD * 2);
  unsigned char* idxb = (unsigned char*)A((size_t)BB * NN * NN);
  unsigned char* hmsg = (unsigned char*)A(BN);
  float* kp = (float*)A(49 * MSG * 4);
  float* vp = (float*)A(49 * MSG * 4);
  size_t scratch0 = off;
  // encoder scratch (dead after latent is built)
  f16* col1 = (f16*)A((size_t)BN * 25 * 96 * 2);
  f16* act1 = (f16*)A((size_t)BN * 25 * 64 * 2);
  f16* col2 = (f16*)A((size_t)BN * 9 * 576 * 2);
  f16* act2 = (f16*)A((size_t)BN * 9 * 128 * 2);
  f16* col3 = (f16*)A((size_t)BN * 1152 * 2);
  // GRU / comm scratch overlays the encoder scratch (used strictly after)
  off = scratch0;
  float* gi = (float*)A((size_t)BN * 768 * 4);
  float* gh = (float*)A((size_t)BN * 768 * 4);
  f16* hn16 = (f16*)A((size_t)BN * HD * 2);
  float* q32 = (float*)A((size_t)BN * MSG * 4);
  f16* q16 = (f16*)A((size_t)BN * MSG * 2);
  f16* kh16 = (f16*)A((size_t)BN * MSG * 2);
  float* vh32 = (float*)A((size_t)BN * MSG * 4);
  float* qkp = (float*)A((size_t)BN * NH * 49 * 4);
  float* sbuf = (float*)A((size_t)BB * NH * NN * NN * 4);
  float* ag32 = (float*)A((size_t)BN * MSG * 4);
  f16* ag16 = (f16*)A((size_t)BN * MSG * 2);
  f16* ag216 = (f16*)A((size_t)BN * MSG * 2);
  (void)ws_size; (void)n_in; (void)out_size;

  const float iscale = 0.17677669529663687f;  // 1/sqrt(MD)
  auto cvt = [&](const float* s, f16* d, int rows, int sk, int sld, int dk) {
    k_cvt<<<((size_t)rows * dk + 255) / 256, 256, 0, stream>>>(s, d, rows, sk, sld, dk);
  };

  // ---- weight conversion to f16 ------------------------------------------
  cvt(convW[0], wW0, 64, 72, 72, 96);
  cvt(convW[1], wW1, 128, 576, 576, 576);
  cvt(convW[2], wW2, 256, 1152, 1152, 1152);
  cvt(recWih, wRih, 768, 261, 261, 288);
  cvt(recWhh, wRhh, 768, 256, 256, 256);
  for (int l = 0; l < 2; l++) {
    cvt(L[l].Wq, wQ[l], 128, 256, 256, 256);
    cvt(L[l].Wk, wKh[l], 128, 256, 272, 256);  // hidden part of Wk
    cvt(L[l].Wv, wVh[l], 128, 256, 272, 256);  // hidden part of Wv
    cvt(L[l].Wagg, wAg[l], 128, 128, 128, 128);
    cvt(L[l].Wih, wCih[l], 768, 128, 128, 128);
    cvt(L[l].Whh, wChh[l], 768, 256, 256, 256);
  }
  cvt(hidden_in, hid16, BN, 256, 256, 256);
  k_relidx<<<(BB * NN * NN + 255) / 256, 256, 0, stream>>>(relpos, idxb);

  // ---- conv encoder (im2col + WMMA GEMM, leaky fused) ---------------------
  k_im2col1<<<((size_t)BN * 25 * 96 + 255) / 256, 256, 0, stream>>>(obs, col1);
  gemm(stream, col1, wW0, convB[0], nullptr, act1, BN * 25, 64, 96, 96, 96, 0, 64, 1.0f, 1);
  k_im2col2<<<((size_t)BN * 9 * 576 + 255) / 256, 256, 0, stream>>>(act1, col2);
  gemm(stream, col2, wW1, convB[1], nullptr, act2, BN * 9, 128, 576, 576, 576, 0, 128, 1.0f, 1);
  k_im2col3<<<((size_t)BN * 1152 + 255) / 256, 256, 0, stream>>>(act2, col3);
  gemm(stream, col3, wW2, convB[2], nullptr, lat16, BN, 256, 1152, 1152, 1152, 0, 288, 1.0f, 1);
  k_latent_tail<<<(BN * 32 + 255) / 256, 256, 0, stream>>>(last_act, lat16);

  // ---- recurrent GRU -------------------------------------------------------
  gemm(stream, lat16, wRih, recBih, gi, nullptr, BN, 768, 288, 288, 288, 768, 0, 1.0f, 0);
  gemm(stream, hid16, wRhh, recBhh, gh, nullptr, BN, 768, 256, 256, 256, 768, 0, 1.0f, 0);
  k_gru<<<(BN * HD + 255) / 256, 256, 0, stream>>>(gi, gh, hidden_in, nullptr, hA32, hA16, BN);

  // ---- two communication layers -------------------------------------------
  auto comm = [&](int l, float* hin32, f16* hin16, float* ho32, f16* ho16, int transp) {
    const CommW& c = L[l];
    k_hasmsg<<<(BN + 255) / 256, 256, 0, stream>>>(cm, hmsg, transp);
    k_layernorm<<<(BN + 7) / 8, 256, 0, stream>>>(hin32, c.ln_s, c.ln_b, hn16, BN);
    gemm(stream, hn16, wQ[l], c.bq, q32, q16, BN, MSG, 256, 256, 256, MSG, MSG, 1.0f, 0);
    gemm(stream, hn16, wKh[l], c.bk, nullptr, kh16, BN, MSG, 256, 256, 256, 0, MSG, 1.0f, 0);
    gemm(stream, hn16, wVh[l], c.bv, vh32, nullptr, BN, MSG, 256, 256, 256, MSG, 0, 1.0f, 0);
    k_petable<<<(49 * MSG + 255) / 256, 256, 0, stream>>>(c.Wk, c.posW, c.posb, kp);
    k_petable<<<(49 * MSG + 255) / 256, 256, 0, stream>>>(c.Wv, c.posW, c.posb, vp);
    k_qkp<<<((size_t)BN * NH * 49 + 255) / 256, 256, 0, stream>>>(q32, kp, qkp, iscale);
    // batched scores S(b,h,j,i) = iscale * q(b,j,h)·kh(b,i,h): 32 batches 128x128x32
    gemm(stream, q16, kh16, nullptr, sbuf, nullptr, NN, NN, MD, MSG, MSG, NN, 0,
         iscale, 0, BB * NH, NH,
         (long long)NN * MSG, MD,           // A: per-b, per-h
         (long long)NN * MSG, MD,           // B: per-b, per-h
         (long long)NH * NN * NN, (long long)NN * NN);  // C: per-b, per-h
    k_softmax<<<(BB * NH * NN + 7) / 8, 256, 0, stream>>>(sbuf, qkp, idxb, cm, transp);
    k_agg<<<(BB * NH * NN + 7) / 8, 256, 0, stream>>>(sbuf, vh32, vp, idxb, ag32, ag16);
    gemm(stream, ag16, wAg[l], c.bagg, nullptr, ag216, BN, MSG, 128, 128, 128, 0, MSG, 1.0f, 0);
    gemm(stream, ag216, wCih[l], c.bih, gi, nullptr, BN, 768, 128, 128, 128, 768, 0, 1.0f, 0);
    gemm(stream, hin16, wChh[l], c.bhh, gh, nullptr, BN, 768, 256, 256, 256, 768, 0, 1.0f, 0);
    k_gru<<<(BN * HD + 255) / 256, 256, 0, stream>>>(gi, gh, hin32, hmsg, ho32, ho16, BN);
  };
  comm(0, hA32, hA16, hB32, hB16, 0);  // req: mask as-is
  comm(1, hB32, hB16, hA32, hA16, 1);  // rep: transposed mask

  // ---- dueling heads -------------------------------------------------------
  k_heads<<<(BN + 7) / 8, 256, 0, stream>>>(hA32, advW, advB, stW, stB, out);
}
